// MultiheadSSM_34265249088356
// MI455X (gfx1250) — matrix operand order repaired and checked by reading, not compiled, not run on hardware
//
#include <hip/hip_runtime.h>
#include <hip/hip_bf16.h>

typedef __attribute__((ext_vector_type(16))) __bf16 v16bf;
typedef __attribute__((ext_vector_type(8)))  __bf16 v8bf;
typedef __attribute__((ext_vector_type(8)))  float  v8f;
typedef __attribute__((ext_vector_type(4)))  float  v4f;

#define B_   8
#define L_   4096
#define C_   256
#define H_   8
#define DI_  512
#define T_   64
#define N_   16
#define R_   16
#define BH_  (B_ * H_)          // 64
#define M1_  (B_ * L_)          // 32768  rows for in/out projections
#define M2_  (BH_ * L_)         // 262144 rows for x_dbl/delta (bh*L)
#define K48_ 48                 // R + 2N

#define LDSF_STRIDE 260         // padded fp32 LDS row (1040 B -> bank-conflict free)
#define LDSB_STRIDE 520         // padded bf16 LDS row (1040 B)

// ---------------------------------------------------------------------------
// CDNA5 async global->LDS copy (ASYNCcnt path), GV addressing mode:
//   copies global memory to the LDS byte offset in vdst, 16 bytes per lane, no VGPR round-trip.
// ---------------------------------------------------------------------------
__device__ __forceinline__ void async_copy_b128(unsigned lds_byte_off, const void* gaddr) {
  asm volatile("global_load_async_to_lds_b128 %0, %1, off"
               :: "v"(lds_byte_off), "v"(gaddr)
               : "memory");
}
__device__ __forceinline__ void wait_async0() {
  asm volatile("s_wait_asynccnt 0x0" ::: "memory");
}

// ---------------------------------------------------------------------------
// Small prep kernels: fp32 -> bf16 weight conversion, A = -exp(A_log)
// ---------------------------------------------------------------------------
__global__ __launch_bounds__(256) void k_f32_to_bf16(const float* __restrict__ in,
                                                     __bf16* __restrict__ out, int n) {
  int i = blockIdx.x * 256 + threadIdx.x;
  if (i < n) out[i] = (__bf16)in[i];
}

__global__ __launch_bounds__(256) void k_negexp(const float* __restrict__ in,
                                                float* __restrict__ out, int n) {
  int i = blockIdx.x * 256 + threadIdx.x;
  if (i < n) out[i] = -__expf(in[i]);
}

// ---------------------------------------------------------------------------
// in_proj: xz = X(32768x256) @ W^T(256x1024).
// Block = 128 threads (4 waves), tile = 16 rows x 256 cols.
// A panel (16x256 fp32) staged once via async-to-LDS; each wave computes a
// 16x64 strip (4 N-tiles), reusing one A fragment across 4 WMMAs per K-step.
// ---------------------------------------------------------------------------
__global__ __launch_bounds__(128) void k_in_proj(const float* __restrict__ X,
                                                 const __bf16* __restrict__ Wb,
                                                 float* __restrict__ u,
                                                 float* __restrict__ zg) {
  __shared__ float sA[16 * LDSF_STRIDE];

  const int tid  = threadIdx.x;
  const int wave = tid >> 5;
  const int lane = tid & 31;
  const int tm   = blockIdx.y;          // 0..2047  (M tiles)
  const int nblk = blockIdx.x;          // 0..3     (256-col super-tiles)

  // ---- stage A block 16x256 fp32 (1024 x b128 chunks, 8 per thread) ----
  const float* Ablk = X + (size_t)tm * 16 * C_;
  for (int c = tid; c < 1024; c += 128) {
    int r = c >> 6;                     // row 0..15
    int col = (c & 63) * 4;             // fp32 column 0..252
    async_copy_b128((unsigned)(size_t)&sA[r * LDSF_STRIDE + col],
                    Ablk + (size_t)r * C_ + col);
  }
  wait_async0();
  __syncthreads();

  const int ml    = lane & 15;
  const int half  = lane >> 4;
  const int kb    = half * 8;
  const int tile0 = nblk * 16 + wave * 4;   // first of this wave's 4 N-tiles

  v8f acc[4] = {};
#pragma unroll
  for (int kk = 0; kk < C_; kk += 32) {
    // A fragment from LDS (ds_load_b128 x4), fp32->bf16 packed in-register
    v4f a0 = *(const v4f*)(&sA[ml * LDSF_STRIDE + kk + kb]);
    v4f a1 = *(const v4f*)(&sA[ml * LDSF_STRIDE + kk + kb + 4]);
    v4f a2 = *(const v4f*)(&sA[ml * LDSF_STRIDE + kk + 16 + kb]);
    v4f a3 = *(const v4f*)(&sA[ml * LDSF_STRIDE + kk + 16 + kb + 4]);
    v16bf a;
#pragma unroll
    for (int j = 0; j < 4; ++j) {
      a[j]      = (__bf16)a0[j];
      a[4 + j]  = (__bf16)a1[j];
      a[8 + j]  = (__bf16)a2[j];
      a[12 + j] = (__bf16)a3[j];
    }
#pragma unroll
    for (int i = 0; i < 4; ++i) {
      const __bf16* brow = Wb + (size_t)((tile0 + i) * 16 + ml) * C_;
      v8bf b0 = *(const v8bf*)(brow + kk + half * 8);
      v8bf b1 = *(const v8bf*)(brow + kk + 16 + half * 8);
      v16bf b;
#pragma unroll
      for (int j = 0; j < 8; ++j) { b[j] = b0[j]; b[8 + j] = b1[j]; }
      acc[i] = __builtin_amdgcn_wmma_f32_16x16x32_bf16(false, a, false, b, (short)0,
                                                       acc[i], false, false);
    }
  }

  // D scatter: VGPR r -> M = r + 8*half, N = lane&15; route to u / zg layouts
#pragma unroll
  for (int i = 0; i < 4; ++i) {
#pragma unroll
    for (int r = 0; r < 8; ++r) {
      int gm = tm * 16 + r + half * 8;
      int gc = (tile0 + i) * 16 + ml;
      int bb = gm >> 12;
      int l  = gm & (L_ - 1);
      if (gc < DI_) {
        int hh = gc >> 6, tt = gc & (T_ - 1);
        u[((size_t)(bb * H_ + hh) * L_ + l) * T_ + tt] = acc[i][r];
      } else {
        int gc2 = gc - DI_;
        int hh = gc2 >> 6, tt = gc2 & (T_ - 1);
        zg[((size_t)(bb * H_ + hh) * L_ + l) * T_ + tt] = acc[i][r];
      }
    }
  }
}

// ---------------------------------------------------------------------------
// x_dbl = U(262144x64) @ x_proj_w^T(64x48): one wave per 16x48 strip,
// A fragment reused across the 3 N-tiles.
// ---------------------------------------------------------------------------
__global__ __launch_bounds__(32) void k_xdbl(const float* __restrict__ U,
                                             const __bf16* __restrict__ Wb,
                                             float* __restrict__ xdbl) {
  const int lane = threadIdx.x;
  const int tm   = blockIdx.x;          // 0..16383
  const int ml   = lane & 15;
  const int half = lane >> 4;
  const int kb   = half * 8;

  const float* arow = U + (size_t)(tm * 16 + ml) * T_;

  v8f acc[3] = {};
#pragma unroll
  for (int kk = 0; kk < T_; kk += 32) {
    v4f a0 = *(const v4f*)(arow + kk + kb);
    v4f a1 = *(const v4f*)(arow + kk + kb + 4);
    v4f a2 = *(const v4f*)(arow + kk + 16 + kb);
    v4f a3 = *(const v4f*)(arow + kk + 16 + kb + 4);
    v16bf a;
#pragma unroll
    for (int j = 0; j < 4; ++j) {
      a[j]      = (__bf16)a0[j];
      a[4 + j]  = (__bf16)a1[j];
      a[8 + j]  = (__bf16)a2[j];
      a[12 + j] = (__bf16)a3[j];
    }
#pragma unroll
    for (int i = 0; i < 3; ++i) {
      const __bf16* brow = Wb + (size_t)(i * 16 + ml) * T_;
      v8bf b0 = *(const v8bf*)(brow + kk + half * 8);
      v8bf b1 = *(const v8bf*)(brow + kk + 16 + half * 8);
      v16bf b;
#pragma unroll
      for (int j = 0; j < 8; ++j) { b[j] = b0[j]; b[8 + j] = b1[j]; }
      acc[i] = __builtin_amdgcn_wmma_f32_16x16x32_bf16(false, a, false, b, (short)0,
                                                       acc[i], false, false);
    }
  }
#pragma unroll
  for (int i = 0; i < 3; ++i) {
#pragma unroll
    for (int r = 0; r < 8; ++r) {
      int gm = tm * 16 + r + half * 8;
      xdbl[(size_t)gm * K48_ + i * 16 + ml] = acc[i][r];
    }
  }
}

// ---------------------------------------------------------------------------
// delta[m][t] = softplus(xdbl[m][0:16] . dt_proj_w[t][:] + dt_proj_b[t])
// ---------------------------------------------------------------------------
__global__ __launch_bounds__(256) void k_delta(const float* __restrict__ xdbl,
                                               const float* __restrict__ dtw,
                                               const float* __restrict__ dtb,
                                               float* __restrict__ delta) {
  int idx = blockIdx.x * 256 + threadIdx.x;   // covers M2_*T_ exactly
  int m = idx >> 6;
  int t = idx & (T_ - 1);
  const float* xr = xdbl + (size_t)m * K48_;
  const float* wr = dtw + t * R_;
  float acc = dtb[t];
#pragma unroll
  for (int r = 0; r < R_; r += 4) {
    v4f xv = *(const v4f*)(xr + r);
    v4f wv = *(const v4f*)(wr + r);
    acc += xv[0] * wv[0] + xv[1] * wv[1] + xv[2] * wv[2] + xv[3] * wv[3];
  }
  delta[idx] = (acc > 20.f) ? acc : __logf(1.f + __expf(acc));
}

// ---------------------------------------------------------------------------
// Selective scan: one block per (b*h), 256 threads = 64 t-rows x 4 lanes,
// each lane holding 4 of the 16 n-states in registers. y reduced over n via
// two wave32 xor-shuffles; SiLU gating fused; output written bf16 into the
// (B,L,DI) layout consumed by out_proj.
// ---------------------------------------------------------------------------
__global__ __launch_bounds__(256) void k_scan(const float* __restrict__ delta,
                                              const float* __restrict__ U,
                                              const float* __restrict__ zg,
                                              const float* __restrict__ xdbl,
                                              const float* __restrict__ Aarr,
                                              const float* __restrict__ Dp,
                                              __bf16* __restrict__ gout) {
  const int bh  = blockIdx.x;           // 0..63
  const int tid = threadIdx.x;
  const int t   = tid >> 2;             // 0..63
  const int gq  = tid & 3;              // lane group within t-row
  const int n0  = gq * 4;
  const int bb  = bh >> 3, hh = bh & 7;

  float At[4];
#pragma unroll
  for (int j = 0; j < 4; ++j) At[j] = Aarr[t * N_ + n0 + j];
  const float Dt = Dp[t];

  const float* drow = delta + (size_t)bh * L_ * T_;
  const float* urow = U     + (size_t)bh * L_ * T_;
  const float* zrow = zg    + (size_t)bh * L_ * T_;
  const float* xrow = xdbl  + (size_t)bh * L_ * K48_;
  __bf16* grow = gout + (size_t)bb * L_ * DI_ + hh * T_ + t;

  float h0 = 0.f, h1 = 0.f, h2 = 0.f, h3 = 0.f;
  for (int l = 0; l < L_; ++l) {
    if ((l & 7) == 0 && l + 8 < L_) {
      __builtin_prefetch(drow + (size_t)(l + 8) * T_ + t, 0, 0);
      __builtin_prefetch(xrow + (size_t)(l + 8) * K48_ + R_ + n0, 0, 0);
    }
    float dlt = drow[l * T_ + t];
    float ut  = urow[l * T_ + t];
    v4f Bv = *(const v4f*)(xrow + (size_t)l * K48_ + R_ + n0);        // B columns
    v4f Cv = *(const v4f*)(xrow + (size_t)l * K48_ + R_ + N_ + n0);   // C columns
    float du = dlt * ut;
    h0 = __expf(dlt * At[0]) * h0 + du * Bv[0];
    h1 = __expf(dlt * At[1]) * h1 + du * Bv[1];
    h2 = __expf(dlt * At[2]) * h2 + du * Bv[2];
    h3 = __expf(dlt * At[3]) * h3 + du * Bv[3];
    float ys = h0 * Cv[0] + h1 * Cv[1] + h2 * Cv[2] + h3 * Cv[3];
    ys += __shfl_xor(ys, 1, 32);
    ys += __shfl_xor(ys, 2, 32);
    if (gq == 0) {
      float zv   = zrow[l * T_ + t];
      float gate = zv / (1.f + __expf(-zv));   // z * sigmoid(z)
      grow[(size_t)l * DI_] = (__bf16)((ys + ut * Dt) * gate);
    }
  }
}

// ---------------------------------------------------------------------------
// out_proj: out = G(32768x512 bf16) @ Wout^T(512x256).
// Block = 128 threads (4 waves), tile = 16 rows x 256 cols (all of N).
// A panel (16x512 bf16) async-staged to LDS once; each wave computes a 16x64
// strip (4 N-tiles) -> 64 WMMAs per wave.
// ---------------------------------------------------------------------------
__global__ __launch_bounds__(128) void k_out_proj(const __bf16* __restrict__ G,
                                                  const __bf16* __restrict__ Wb,
                                                  float* __restrict__ out) {
  __shared__ __bf16 sG[16 * LDSB_STRIDE];

  const int tid  = threadIdx.x;
  const int wave = tid >> 5;
  const int lane = tid & 31;
  const int tm   = blockIdx.x;          // 0..2047

  // ---- stage A block 16x512 bf16 (1024 x b128 chunks, 8 per thread) ----
  const __bf16* Gblk = G + (size_t)tm * 16 * DI_;
  for (int c = tid; c < 1024; c += 128) {
    int r = c >> 6;                     // row 0..15
    int col = (c & 63) * 8;             // bf16 column 0..504
    async_copy_b128((unsigned)(size_t)&sG[r * LDSB_STRIDE + col],
                    Gblk + (size_t)r * DI_ + col);
  }
  wait_async0();
  __syncthreads();

  const int ml    = lane & 15;
  const int half  = lane >> 4;
  const int kb    = half * 8;
  const int tile0 = wave * 4;           // 16 N-tiles split 4-per-wave

  v8f acc[4] = {};
#pragma unroll
  for (int kk = 0; kk < DI_; kk += 32) {
    v8bf a0 = *(const v8bf*)(&sG[ml * LDSB_STRIDE + kk + kb]);
    v8bf a1 = *(const v8bf*)(&sG[ml * LDSB_STRIDE + kk + 16 + kb]);
    v16bf a;
#pragma unroll
    for (int j = 0; j < 8; ++j) { a[j] = a0[j]; a[8 + j] = a1[j]; }
#pragma unroll
    for (int i = 0; i < 4; ++i) {
      const __bf16* brow = Wb + (size_t)((tile0 + i) * 16 + ml) * DI_;
      v8bf b0 = *(const v8bf*)(brow + kk + half * 8);
      v8bf b1 = *(const v8bf*)(brow + kk + 16 + half * 8);
      v16bf b;
#pragma unroll
      for (int j = 0; j < 8; ++j) { b[j] = b0[j]; b[8 + j] = b1[j]; }
      acc[i] = __builtin_amdgcn_wmma_f32_16x16x32_bf16(false, a, false, b, (short)0,
                                                       acc[i], false, false);
    }
  }
#pragma unroll
  for (int i = 0; i < 4; ++i) {
#pragma unroll
    for (int r = 0; r < 8; ++r) {
      int gm = tm * 16 + r + half * 8;
      out[(size_t)gm * C_ + (tile0 + i) * 16 + ml] = acc[i][r];
    }
  }
}

// ---------------------------------------------------------------------------
extern "C" void kernel_launch(void* const* d_in, const int* in_sizes, int n_in,
                              void* d_out, int out_size, void* d_ws, size_t ws_size,
                              hipStream_t stream) {
  (void)in_sizes; (void)n_in; (void)out_size; (void)ws_size;

  const float* inputs  = (const float*)d_in[0];   // (8,4096,256)
  const float* in_w    = (const float*)d_in[1];   // (1024,256)
  const float* out_w   = (const float*)d_in[2];   // (256,512)
  const float* xproj_w = (const float*)d_in[3];   // (48,64)
  const float* dt_w    = (const float*)d_in[4];   // (64,16)
  const float* dt_b    = (const float*)d_in[5];   // (64,)
  const float* A_log   = (const float*)d_in[6];   // (64,16)
  const float* Dparam  = (const float*)d_in[7];   // (64,)
  float* out = (float*)d_out;                     // (8,4096,256)

  char* ws = (char*)d_ws;
  size_t off = 0;
  auto alloc = [&](size_t bytes) -> void* {
    void* p = (void*)(ws + off);
    off += bytes;
    off = (off + 255) & ~(size_t)255;
    return p;
  };

  __bf16* wb_in  = (__bf16*)alloc((size_t)1024 * 256 * 2);
  __bf16* wb_out = (__bf16*)alloc((size_t)256 * 512 * 2);
  __bf16* wb_xp  = (__bf16*)alloc((size_t)K48_ * T_ * 2);
  float*  Aarr   = (float*) alloc((size_t)T_ * N_ * 4);
  float*  u      = (float*) alloc((size_t)M2_ * T_ * 4);     // 67 MB
  float*  zgbuf  = (float*) alloc((size_t)M2_ * T_ * 4);     // 67 MB
  float*  deltab = (float*) alloc((size_t)M2_ * T_ * 4);     // 67 MB
  float*  xdbl   = (float*) alloc((size_t)M2_ * K48_ * 4);   // 50 MB
  __bf16* gbuf   = (__bf16*)alloc((size_t)M1_ * DI_ * 2);    // 33 MB

  // weight prep
  k_f32_to_bf16<<<(1024 * 256 + 255) / 256, 256, 0, stream>>>(in_w, wb_in, 1024 * 256);
  k_f32_to_bf16<<<(256 * 512 + 255) / 256, 256, 0, stream>>>(out_w, wb_out, 256 * 512);
  k_f32_to_bf16<<<(K48_ * T_ + 255) / 256, 256, 0, stream>>>(xproj_w, wb_xp, K48_ * T_);
  k_negexp<<<(T_ * N_ + 255) / 256, 256, 0, stream>>>(A_log, Aarr, T_ * N_);

  // pipeline
  k_in_proj <<<dim3(4, M1_ / 16), 128, 0, stream>>>(inputs, wb_in, u, zgbuf);
  k_xdbl    <<<M2_ / 16, 32, 0, stream>>>(u, wb_xp, xdbl);
  k_delta   <<<(M2_ * T_) / 256, 256, 0, stream>>>(xdbl, dt_w, dt_b, deltab);
  k_scan    <<<BH_, 256, 0, stream>>>(deltab, u, zgbuf, xdbl, Aarr, Dparam, gbuf);
  k_out_proj<<<M1_ / 16, 128, 0, stream>>>(gbuf, wb_out, out);
}
